// otias_c_x4_65755949302329
// MI455X (gfx1250) — compile-verified, matrix operand order checked by reading, not run on hardware
//
#include <hip/hip_runtime.h>
#include <hip/hip_bf16.h>
#include <cstdint>

typedef _Float16 h16;
typedef __attribute__((ext_vector_type(16))) _Float16 v16h;
typedef __attribute__((ext_vector_type(8)))  _Float16 v8h;
typedef __attribute__((ext_vector_type(8)))  float    v8f;
typedef unsigned int u32x4 __attribute__((ext_vector_type(4)));
typedef int          i32x8 __attribute__((ext_vector_type(8)));
typedef int          i32x4 __attribute__((ext_vector_type(4)));

__device__ __forceinline__ v8f wmma16(v16h a, v16h b, v8f c) {
  // D = A(16x32 f16) * B(32x16 f16) + C(16x16 f32)
  return __builtin_amdgcn_wmma_f32_16x16x32_f16(false, a, false, b, (short)0, c,
                                                false, false);
}

__device__ __forceinline__ int clampi(int v, int lo, int hi) {
  return v < lo ? lo : (v > hi ? hi : v);
}

// Issue one 1-D TDM tile load (nelem f16, contiguous) into LDS byte offset la.
// Caller must barrier before consuming (wait is done here by the issuing wave).
__device__ __forceinline__ void tdm_load_1d(const h16* src, unsigned la,
                                            unsigned nelem) {
#if defined(__gfx1250__) && __has_builtin(__builtin_amdgcn_tensor_load_to_lds)
  unsigned long long ga = (unsigned long long)(uintptr_t)src;
  u32x4 g0 = {};
  g0[0] = 1u;                                           // count=1 (user D#)
  g0[1] = la;                                           // lds_addr (bytes)
  g0[2] = (unsigned)ga;                                 // global_addr[31:0]
  g0[3] = (unsigned)((ga >> 32) & 0x1ffffffu) | 0x80000000u;  // hi | type=2
  i32x8 g1 = {};
  g1[0] = (int)(1u << 16);                              // data_size = 2B
  g1[1] = (int)((nelem & 0xffffu) << 16);               // tensor_dim0 lo16
  g1[2] = (int)((nelem >> 16) & 0xffffu) | (1 << 16);   // dim0 hi | dim1=1
  g1[3] = (int)((nelem & 0xffffu) << 16);               // tile_dim0 (<=65535)
  g1[4] = 1;                                            // tile_dim1 = 1
  g1[5] = (int)nelem;                                   // dim0_stride lo32
  i32x4 z4 = {};
#if __clang_major__ >= 23
  i32x8 z8 = {};
  __builtin_amdgcn_tensor_load_to_lds(g0, g1, z4, z4, z8, 0);
#else
  __builtin_amdgcn_tensor_load_to_lds(g0, g1, z4, z4, 0);
#endif
  __builtin_amdgcn_s_wait_tensorcnt((short)0);
#else
  (void)src; (void)la; (void)nelem;
#endif
}

// ---------------------------------------------------------------------------
// Weight packing: conv OIHW f32 -> WMMA A-fragment order, K = s*Cpad + ci
// dst[((mt*KT + kt)*32 + lane)*16 + j]; lane<16: M=lane, K lo half; lane>=16: hi
// ---------------------------------------------------------------------------
__global__ void k_pack_conv_w(const float* __restrict__ w, h16* __restrict__ dst,
                              int Cout, int Cin, int Cpad, int KT) {
  int t = blockIdx.x * 256 + threadIdx.x;
  int total = (Cout / 16) * KT * 512;
  if (t >= total) return;
  int j = t & 15, lane = (t >> 4) & 31, kt = (t >> 9) % KT, mt = t / (KT * 512);
  int co = mt * 16 + (lane & 15);
  int k = kt * 32 + (lane >> 4) * 16 + j;
  int s = k / Cpad, ci = k % Cpad;
  int ky = s / 3, kx = s % 3;
  float v = (ci < Cin) ? w[((size_t)(co * Cin + ci) * 3 + ky) * 3 + kx] : 0.f;
  dst[t] = (h16)v;
}

// GEMM weights: src W[Kin][Nin] row-major f32 -> packed [ft][kt][lane][16]
__global__ void k_pack_gemm_w(const float* __restrict__ w, h16* __restrict__ dst,
                              int Kin, int Nin, int KT, int FT) {
  int t = blockIdx.x * 256 + threadIdx.x;
  int total = FT * KT * 512;
  if (t >= total) return;
  int j = t & 15, lane = (t >> 4) & 31, kt = (t >> 9) % KT, ft = t / (KT * 512);
  int f = ft * 16 + (lane & 15);
  int k = kt * 32 + (lane >> 4) * 16 + j;
  float v = (k < Kin && f < Nin) ? w[(size_t)k * Nin + f] : 0.f;
  dst[t] = (h16)v;
}

// ---------------------------------------------------------------------------
// Input prep: NCHW f32 -> NHWC f16 with channel padding
// ---------------------------------------------------------------------------
__global__ void k_prep_spa(const float* __restrict__ msi, const float* __restrict__ lms,
                           h16* __restrict__ dst) {
  size_t t = (size_t)blockIdx.x * 256 + threadIdx.x;  // 16*4096*64
  if (t >= (size_t)16 * 4096 * 64) return;
  int c = (int)(t & 63);
  size_t pq = t >> 6;
  int q = (int)(pq % 4096), b = (int)(pq / 4096);
  float v = 0.f;
  if (c < 3)       v = msi[((size_t)b * 3 + c) * 4096 + q];
  else if (c < 34) v = lms[((size_t)b * 31 + (c - 3)) * 4096 + q];
  dst[t] = (h16)v;
}

__global__ void k_prep_spe(const float* __restrict__ x, h16* __restrict__ dst) {
  size_t t = (size_t)blockIdx.x * 256 + threadIdx.x;  // 16*256*64
  if (t >= (size_t)16 * 256 * 64) return;
  int c = (int)(t & 63);
  size_t pq = t >> 6;
  int q = (int)(pq % 256), b = (int)(pq / 256);
  float v = (c < 31) ? x[((size_t)b * 31 + c) * 256 + q] : 0.f;
  dst[t] = (h16)v;
}

// ---------------------------------------------------------------------------
// 3x3 SAME conv as implicit GEMM, templated on padded input channels.
// Block = 256 threads = 8 waves; wave m computes output channels [16m,16m+16)
// for a shared 4x4 pixel tile. Input 6x6xCPAD window staged in LDS (zero-fill
// borders); pre-packed weight fragments staged in LDS via the TDM.
// ---------------------------------------------------------------------------
template <int CPAD>
__global__ __launch_bounds__(256) void k_conv3x3_wmma(
    const h16* __restrict__ in, const h16* __restrict__ wp,
    const float* __restrict__ bias, const h16* __restrict__ addend,
    h16* __restrict__ out, int H, int W, int relu) {
  constexpr int CT = CPAD / 32;       // K-tiles per kernel tap
  constexpr int KT = 9 * CT;          // total K-tiles
  constexpr int INW = 36 * CPAD;      // input window h16 elements
  constexpr int CP8 = CPAD / 8;
  extern __shared__ __attribute__((aligned(32))) h16 smem[];

  const int tid = threadIdx.x;
  const int lane = tid & 31;
  const int mt = tid >> 5;            // wave id == cout tile
  const int tile = blockIdx.x, b = blockIdx.z;
  const int tw = W >> 2;
  const int ty = tile / tw, tx = tile % tw;

  // ---- stage 6x6xCPAD input window into LDS (borders zero-filled) ----
  const h16* inb = in + (size_t)b * H * W * CPAD;
  for (int i = tid; i < 36 * CP8; i += 256) {
    int pix = i / CP8;
    int co = (i % CP8) * 8;
    int y = ty * 4 + pix / 6 - 1;
    int x = tx * 4 + pix % 6 - 1;
    v8h v = {};
    if (y >= 0 && y < H && x >= 0 && x < W)
      v = *(const v8h*)(inb + ((size_t)y * W + x) * CPAD + co);
    *(v8h*)(smem + pix * CPAD + co) = v;
  }

  // ---- stage this wave's weight fragments via TDM ----
  h16* wlds = smem + INW;  // 8 * KT * 512 h16
#if defined(__gfx1250__) && __has_builtin(__builtin_amdgcn_tensor_load_to_lds)
  {
    unsigned mtu = (unsigned)__builtin_amdgcn_readfirstlane(mt);
    tdm_load_1d(wp + (size_t)mtu * KT * 512,
                (unsigned)(INW * 2) + mtu * (unsigned)(KT * 1024),
                (unsigned)(KT * 512));
  }
#else
  for (int i = tid; i < KT * 512; i += 256)  // 8*KT*512 h16 in v8h chunks
    *(v8h*)(wlds + (size_t)i * 8) = *(const v8h*)(wp + (size_t)i * 8);
#endif
  __syncthreads();

  // ---- implicit GEMM: 9 taps x CT channel tiles, all offsets constant ----
  // Fragments for each tap are preloaded into registers so ds loads clause up
  // and the WMMA chain issues without per-op dscnt waits.
  const int n = lane & 15, kh = lane >> 4;
  const int bpy = n >> 2, bpx = n & 3;
  const h16* aw = wlds + (size_t)mt * KT * 512 + (size_t)lane * 16;
  v8f acc = {};
#pragma unroll
  for (int s = 0; s < 9; ++s) {
    const int ky = s / 3, kx = s % 3;
    const h16* bbase = smem + ((bpy + ky) * 6 + bpx + kx) * CPAD + kh * 16;
    v16h af[CT], bf[CT];
#pragma unroll
    for (int ct = 0; ct < CT; ++ct) {
      af[ct] = *(const v16h*)(aw + (size_t)(s * CT + ct) * 512);
      bf[ct] = *(const v16h*)(bbase + ct * 32);
    }
#pragma unroll
    for (int ct = 0; ct < CT; ++ct) acc = wmma16(af[ct], bf[ct], acc);
  }

  // ---- epilogue: bias (+addend) (+relu) -> NHWC f16, 8 contig channels ----
  const int py = ty * 4 + bpy, px = tx * 4 + bpx;
  const int co = mt * 16 + kh * 8;
  const size_t ob = (((size_t)b * H + py) * W + px) * 128 + co;
  v8h ad = {};
  if (addend) ad = *(const v8h*)(addend + ob);
  v8h o;
#pragma unroll
  for (int r = 0; r < 8; ++r) {
    float v = acc[r] + bias[co + r];
    if (addend) v += (float)ad[r];
    if (relu) v = v > 0.f ? v : 0.f;
    o[r] = (h16)v;
  }
  *(v8h*)(out + ob) = o;
}

// ---------------------------------------------------------------------------
// WMMA GEMM templated on feature-tile count (branch-free inner loop).
// Block = 256 threads = 8 waves; wave w handles sample tile blockIdx.x*8+w.
// All waves share one LDS copy of the packed weight panel (NFT*KT*512 f16),
// staged by wave 0 with a single TDM descriptor. Weight fragments for each
// K-step are preloaded into registers (one ds clause + one wait per step).
// out[row][f] = act(A[row][:K] @ W[:,f] + bias[f] (+addend))
// act: 0=none, 1=relu, 2=gelu(exact)
// ---------------------------------------------------------------------------
template <int NFT>
__global__ __launch_bounds__(256) void k_gemm_wmma(
    const h16* __restrict__ A, int lda, int KT,
    const h16* __restrict__ wp, const float* __restrict__ bias, int nbias,
    h16* __restrict__ out, int ldc, const h16* __restrict__ addend, int act) {
  extern __shared__ __attribute__((aligned(32))) h16 wl[];
  const int tid = threadIdx.x, lane = tid & 31, wv = tid >> 5;
  const int st = blockIdx.x * 8 + wv;
  const int ftbase = blockIdx.y * NFT;
  const int nwh = NFT * KT * 512;                   // h16 elements in panel
  const h16* wsrc = wp + (size_t)ftbase * KT * 512;
#if defined(__gfx1250__) && __has_builtin(__builtin_amdgcn_tensor_load_to_lds)
  if (wv == 0) tdm_load_1d(wsrc, 0u, (unsigned)nwh);
#else
  for (int i = tid; i < (nwh >> 3); i += 256)
    *(v8h*)(wl + (size_t)i * 8) = *(const v8h*)(wsrc + (size_t)i * 8);
#endif
  __syncthreads();

  const int nn = lane & 15, kh = lane >> 4;
  const int row = st * 16 + nn;
  v8f acc[NFT];
#pragma unroll
  for (int f = 0; f < NFT; ++f) acc[f] = (v8f){};
  const h16* arow = A + (size_t)row * lda + kh * 16;
  for (int kt = 0; kt < KT; ++kt) {
    if (kt + 1 < KT) __builtin_prefetch(arow + (kt + 1) * 32, 0, 1);
    v16h bact = *(const v16h*)(arow + kt * 32);
    v16h aw[NFT];
#pragma unroll
    for (int f = 0; f < NFT; ++f)
      aw[f] = *(const v16h*)(wl + ((size_t)(f * KT + kt) * 32 + lane) * 16);
#pragma unroll
    for (int f = 0; f < NFT; ++f) acc[f] = wmma16(aw[f], bact, acc[f]);
  }
#pragma unroll
  for (int f = 0; f < NFT; ++f) {
    const int fb = (ftbase + f) * 16 + kh * 8;
    const size_t ob = (size_t)row * ldc + fb;
    v8h ad = {};
    if (addend) ad = *(const v8h*)(addend + ob);
    v8h o;
#pragma unroll
    for (int r = 0; r < 8; ++r) {
      float v = acc[f][r];
      int fc = fb + r;
      if (fc < nbias) v += bias[fc];
      if (addend) v += (float)ad[r];
      if (act == 1) v = v > 0.f ? v : 0.f;
      else if (act == 2) v = 0.5f * v * (1.f + erff(v * 0.70710678118f));
      o[r] = (h16)v;
    }
    *(v8h*)(out + ob) = o;
  }
}

// ---------------------------------------------------------------------------
// k x k max pool on NHWC f16 128ch
// ---------------------------------------------------------------------------
__global__ void k_maxpool(const h16* __restrict__ in, int H, int W, int k,
                          h16* __restrict__ out) {
  int c = threadIdx.x;                 // 128
  size_t p = blockIdx.x;               // b*OH*OW
  int OW = W / k, OH = H / k;
  int ox = (int)(p % OW);
  int oy = (int)((p / OW) % OH);
  int b  = (int)(p / ((size_t)OW * OH));
  float m = -3.0e38f;
  for (int dy = 0; dy < k; ++dy)
    for (int dx = 0; dx < k; ++dx) {
      float v = (float)in[(((size_t)b * H + oy * k + dy) * W + ox * k + dx) * 128 + c];
      m = v > m ? v : m;
    }
  out[p * 128 + c] = (h16)m;
}

// ---------------------------------------------------------------------------
// Build level-MLP input rows: [f_s(128) | g_s(128) | gl_s(128) | rel(2) | pad]
// rows = Nq*4 (4 shifted nearest samples), lda = 416 (13 K-tiles)
// ---------------------------------------------------------------------------
__global__ void k_gather(const h16* __restrict__ feat, int fh, int fw,
                         const h16* __restrict__ guide,
                         const h16* __restrict__ guidelr, int glh, int glw,
                         int Hq, int Wq, h16* __restrict__ A) {
  const int r = blockIdx.x;     // q*4 + o
  const int c = threadIdx.x;    // 0..127
  const int q = r >> 2, o = r & 3;
  const int qy = q / Wq, qx = q % Wq;
  const float cy = -1.f + (2.f * qy + 1.f) / (float)Hq;
  const float cx = -1.f + (2.f * qx + 1.f) / (float)Wq;
  const float vx = (o < 2) ? -1.f : 1.f;
  const float vy = (o & 1) ? 1.f : -1.f;
  const float cy_ = cy + vx / (float)fh;
  const float cx_ = cx + vy / (float)fw;
  h16* row = A + (size_t)r * 416;
  // feat sample (nearest, zeros padding)
  float fy = roundf(((cy_ + 1.f) * fh - 1.f) * 0.5f);
  float fx = roundf(((cx_ + 1.f) * fw - 1.f) * 0.5f);
  bool vf = (fy >= 0.f) && (fy < (float)fh) && (fx >= 0.f) && (fx < (float)fw);
  int iy = clampi((int)fy, 0, fh - 1), ix = clampi((int)fx, 0, fw - 1);
  row[c] = vf ? feat[((size_t)iy * fw + ix) * 128 + c] : (h16)0.f;
  // hr guide sample at the un-shifted coord: exact pixel (qy,qx)
  row[128 + c] = guide[((size_t)qy * Wq + qx) * 128 + c];
  // low-res guide at shifted coord
  float gy = roundf(((cy_ + 1.f) * glh - 1.f) * 0.5f);
  float gx = roundf(((cx_ + 1.f) * glw - 1.f) * 0.5f);
  bool vg = (gy >= 0.f) && (gy < (float)glh) && (gx >= 0.f) && (gx < (float)glw);
  int gyi = clampi((int)gy, 0, glh - 1), gxi = clampi((int)gx, 0, glw - 1);
  row[256 + c] = vg ? guidelr[((size_t)gyi * glw + gxi) * 128 + c] : (h16)0.f;
  if (c < 32) {   // rel coords + K padding (cols 384..415)
    float val = 0.f;
    if (c == 0) {
      float csy = vf ? (-1.f + (2.f * iy + 1.f) / (float)fh) : 0.f;
      val = (cy - csy) * (float)fh;
    } else if (c == 1) {
      float csx = vf ? (-1.f + (2.f * ix + 1.f) / (float)fw) : 0.f;
      val = (cx - csx) * (float)fw;
    }
    row[384 + c] = (h16)val;
  }
}

// softmax over 8 candidate preds per (q,c) and weighted sum -> recon
__global__ void k_softmax_recon(const h16* __restrict__ preds,
                                const h16* __restrict__ wgout,
                                h16* __restrict__ recon) {
  const int q = blockIdx.x, c = threadIdx.x;  // 128 threads
  float w[8], m = -3.0e38f;
#pragma unroll
  for (int j = 0; j < 8; ++j) {
    w[j] = (float)wgout[((size_t)q * 8 + j) * 128 + c];
    m = w[j] > m ? w[j] : m;
  }
  float s = 0.f;
#pragma unroll
  for (int j = 0; j < 8; ++j) { w[j] = expf(w[j] - m); s += w[j]; }
  float inv = 1.f / s, acc = 0.f;
#pragma unroll
  for (int j = 0; j < 8; ++j) {
    float p = (float)preds[((size_t)q * 4 + (j >> 1)) * 256 + (j & 1) * 128 + c];
    acc += p * w[j] * inv;
  }
  recon[(size_t)q * 128 + c] = (h16)acc;
}

// out[b] = lms[b] + l2 (31 channels, f32 NCHW)
__global__ void k_final(const float* __restrict__ lms, const h16* __restrict__ l2,
                        float* __restrict__ out, int b) {
  int t = blockIdx.x * 256 + threadIdx.x;  // 31*4096
  if (t >= 31 * 4096) return;
  int ch = t / 4096, q = t % 4096;
  out[(size_t)b * 31 * 4096 + t] =
      lms[(size_t)b * 31 * 4096 + t] + (float)l2[(size_t)q * 32 + ch];
}

// ---------------------------------------------------------------------------
extern "C" void kernel_launch(void* const* d_in, const int* in_sizes, int n_in,
                              void* d_out, int out_size, void* d_ws, size_t ws_size,
                              hipStream_t stream) {
  (void)in_sizes; (void)n_in; (void)out_size; (void)ws_size;
  const float* HR_MSI    = (const float*)d_in[0];
  const float* lms       = (const float*)d_in[1];
  const float* LR_HSI    = (const float*)d_in[2];
  const float* spa_head_w = (const float*)d_in[3];
  const float* spa_head_b = (const float*)d_in[4];
  const float* spa_res_w  = (const float*)d_in[5];
  const float* spa_res_b  = (const float*)d_in[6];
  const float* spa_tail_w = (const float*)d_in[7];
  const float* spa_tail_b = (const float*)d_in[8];
  const float* spe_head_w = (const float*)d_in[9];
  const float* spe_head_b = (const float*)d_in[10];
  const float* spe_res_w  = (const float*)d_in[11];
  const float* spe_res_b  = (const float*)d_in[12];
  const float* spe_tail_w = (const float*)d_in[13];
  const float* spe_tail_b = (const float*)d_in[14];
  const float* l1_w = (const float*)d_in[15];
  const float* l1_b = (const float*)d_in[16];
  const float* wg32_w1 = (const float*)d_in[17];
  const float* wg32_b1 = (const float*)d_in[18];
  const float* wg32_w2 = (const float*)d_in[19];
  const float* wg32_b2 = (const float*)d_in[20];
  const float* ffn32_w1 = (const float*)d_in[21];
  const float* ffn32_b1 = (const float*)d_in[22];
  const float* ffn32_w2 = (const float*)d_in[23];
  const float* ffn32_b2 = (const float*)d_in[24];
  const float* l2_w = (const float*)d_in[25];
  const float* l2_b = (const float*)d_in[26];
  const float* wg64_w1 = (const float*)d_in[27];
  const float* wg64_b1 = (const float*)d_in[28];
  const float* wg64_w2 = (const float*)d_in[29];
  const float* wg64_b2 = (const float*)d_in[30];
  const float* ffn64_w1 = (const float*)d_in[31];
  const float* ffn64_b1 = (const float*)d_in[32];
  const float* ffn64_w2 = (const float*)d_in[33];
  const float* ffn64_b2 = (const float*)d_in[34];

  char* wsp = (char*)d_ws;
  size_t off = 0;
  auto alloc = [&](size_t elems) -> h16* {
    h16* p = (h16*)(wsp + off);
    off = (off + elems * sizeof(h16) + 255) & ~(size_t)255;
    return p;
  };
  auto packc = [&](const float* w, h16* dst, int Cout, int Cin, int Cpad, int KT) {
    int total = (Cout / 16) * KT * 512;
    k_pack_conv_w<<<(total + 255) / 256, 256, 0, stream>>>(w, dst, Cout, Cin, Cpad, KT);
  };
  auto packg = [&](const float* w, h16* dst, int Kin, int Nin, int KT, int FT) {
    int total = FT * KT * 512;
    k_pack_gemm_w<<<(total + 255) / 256, 256, 0, stream>>>(w, dst, Kin, Nin, KT, FT);
  };
  // dispatch helper: rows must be a multiple of 128; FT_total % NFT == 0
  auto gemm = [&](int NFT, int rows, const h16* A, int lda, int KT, const h16* wp,
                  const float* bias, int nbias, h16* out, int ldc,
                  const h16* addend, int FT_total, int act) {
    dim3 g(rows / 128, FT_total / NFT);
    size_t sh = (size_t)NFT * KT * 512 * sizeof(h16);
    if (NFT == 8)
      k_gemm_wmma<8><<<g, 256, sh, stream>>>(A, lda, KT, wp, bias, nbias, out, ldc, addend, act);
    else if (NFT == 4)
      k_gemm_wmma<4><<<g, 256, sh, stream>>>(A, lda, KT, wp, bias, nbias, out, ldc, addend, act);
    else
      k_gemm_wmma<2><<<g, 256, sh, stream>>>(A, lda, KT, wp, bias, nbias, out, ldc, addend, act);
  };

  // -------- pack all weights to WMMA fragment order (f16) --------
  h16* pk_spa_head = alloc(8 * 18 * 512); packc(spa_head_w, pk_spa_head, 128, 34, 64, 18);
  h16* pk_spa_res[8];
  for (int i = 0; i < 8; ++i) {
    pk_spa_res[i] = alloc(8 * 36 * 512);
    packc(spa_res_w + (size_t)i * 128 * 128 * 9, pk_spa_res[i], 128, 128, 128, 36);
  }
  h16* pk_spa_tail = alloc(8 * 36 * 512); packc(spa_tail_w, pk_spa_tail, 128, 128, 128, 36);
  h16* pk_spe_head = alloc(8 * 18 * 512); packc(spe_head_w, pk_spe_head, 128, 31, 64, 18);
  h16* pk_spe_res[8];
  for (int i = 0; i < 8; ++i) {
    pk_spe_res[i] = alloc(8 * 36 * 512);
    packc(spe_res_w + (size_t)i * 128 * 128 * 9, pk_spe_res[i], 128, 128, 128, 36);
  }
  h16* pk_spe_tail = alloc(8 * 36 * 512); packc(spe_tail_w, pk_spe_tail, 128, 128, 128, 36);

  h16* pk_l1   = alloc(16 * 13 * 512); packg(l1_w, pk_l1, 386, 256, 13, 16);
  h16* pk_l2   = alloc(16 * 13 * 512); packg(l2_w, pk_l2, 386, 256, 13, 16);
  h16* pk_wg32_1 = alloc(4 * 4 * 512); packg(wg32_w1, pk_wg32_1, 128, 64, 4, 4);
  h16* pk_wg32_2 = alloc(8 * 2 * 512); packg(wg32_w2, pk_wg32_2, 64, 128, 2, 8);
  h16* pk_ffn32_1 = alloc(4 * 4 * 512); packg(ffn32_w1, pk_ffn32_1, 128, 64, 4, 4);
  h16* pk_ffn32_2 = alloc(8 * 2 * 512); packg(ffn32_w2, pk_ffn32_2, 64, 128, 2, 8);
  h16* pk_wg64_1 = alloc(4 * 4 * 512); packg(wg64_w1, pk_wg64_1, 128, 64, 4, 4);
  h16* pk_wg64_2 = alloc(8 * 2 * 512); packg(wg64_w2, pk_wg64_2, 64, 128, 2, 8);
  h16* pk_ffn64_1 = alloc(4 * 4 * 512); packg(ffn64_w1, pk_ffn64_1, 128, 64, 4, 4);
  h16* pk_ffn64_2 = alloc(2 * 2 * 512); packg(ffn64_w2, pk_ffn64_2, 64, 31, 2, 2);

  const size_t sh64  = (size_t)(36 * 64  + 8 * 18 * 512) * sizeof(h16);
  const size_t sh128 = (size_t)(36 * 128 + 8 * 36 * 512) * sizeof(h16);

  // -------- EDSR spatial branch (NHWC f16, 16x64x64) --------
  h16* x_spa = alloc((size_t)16 * 4096 * 64);
  k_prep_spa<<<(16 * 4096 * 64 + 255) / 256, 256, 0, stream>>>(HR_MSI, lms, x_spa);
  h16* bufX = alloc((size_t)16 * 4096 * 128);
  h16* bufR = alloc((size_t)16 * 4096 * 128);
  h16* bufT = alloc((size_t)16 * 4096 * 128);
  h16* hr_spa = alloc((size_t)16 * 4096 * 128);
  dim3 cg64(256, 1, 16);
  k_conv3x3_wmma<64><<<cg64, 256, sh64, stream>>>(x_spa, pk_spa_head, spa_head_b,
                                                  nullptr, bufX, 64, 64, 0);
  hipMemcpyAsync(bufR, bufX, (size_t)16 * 4096 * 128 * sizeof(h16),
                 hipMemcpyDeviceToDevice, stream);
  for (int i = 0; i < 4; ++i) {
    k_conv3x3_wmma<128><<<cg64, 256, sh128, stream>>>(bufR, pk_spa_res[2 * i],
        spa_res_b + (size_t)i * 256, nullptr, bufT, 64, 64, 1);
    k_conv3x3_wmma<128><<<cg64, 256, sh128, stream>>>(bufT, pk_spa_res[2 * i + 1],
        spa_res_b + (size_t)i * 256 + 128, bufR, bufR, 64, 64, 0);
  }
  k_conv3x3_wmma<128><<<cg64, 256, sh128, stream>>>(bufR, pk_spa_tail, spa_tail_b,
                                                    bufX, hr_spa, 64, 64, 0);

  // -------- EDSR spectral branch (16x16x16) --------
  h16* x_spe = alloc((size_t)16 * 256 * 64);
  k_prep_spe<<<(16 * 256 * 64 + 255) / 256, 256, 0, stream>>>(LR_HSI, x_spe);
  h16* sX = alloc((size_t)16 * 256 * 128);
  h16* sR = alloc((size_t)16 * 256 * 128);
  h16* sT = alloc((size_t)16 * 256 * 128);
  h16* lr_spe = alloc((size_t)16 * 256 * 128);
  dim3 cg16(16, 1, 16);
  k_conv3x3_wmma<64><<<cg16, 256, sh64, stream>>>(x_spe, pk_spe_head, spe_head_b,
                                                  nullptr, sX, 16, 16, 0);
  hipMemcpyAsync(sR, sX, (size_t)16 * 256 * 128 * sizeof(h16),
                 hipMemcpyDeviceToDevice, stream);
  for (int i = 0; i < 4; ++i) {
    k_conv3x3_wmma<128><<<cg16, 256, sh128, stream>>>(sR, pk_spe_res[2 * i],
        spe_res_b + (size_t)i * 256, nullptr, sT, 16, 16, 1);
    k_conv3x3_wmma<128><<<cg16, 256, sh128, stream>>>(sT, pk_spe_res[2 * i + 1],
        spe_res_b + (size_t)i * 256 + 128, sR, sR, 16, 16, 0);
  }
  k_conv3x3_wmma<128><<<cg16, 256, sh128, stream>>>(sR, pk_spe_tail, spe_tail_b,
                                                    sX, lr_spe, 16, 16, 0);

  // -------- guides via max pooling --------
  h16* guide32 = alloc((size_t)16 * 1024 * 128);
  h16* guide16 = alloc((size_t)16 * 256 * 128);
  k_maxpool<<<16 * 1024, 128, 0, stream>>>(hr_spa, 64, 64, 2, guide32);
  k_maxpool<<<16 * 256, 128, 0, stream>>>(hr_spa, 64, 64, 4, guide16);

  // -------- level stages (per batch image; scratch sized for level 2) ------
  h16* l1feat = alloc((size_t)16 * 1024 * 128);
  h16* Abuf  = alloc((size_t)16384 * 416);
  h16* preds = alloc((size_t)16384 * 256);
  h16* wgh   = alloc((size_t)32768 * 64);
  h16* wgo   = alloc((size_t)32768 * 128);
  h16* recon = alloc((size_t)4096 * 128);
  h16* ffnh  = alloc((size_t)4096 * 64);
  h16* l2o   = alloc((size_t)4096 * 32);

  for (int b = 0; b < 16; ++b) {
    const h16* spe_b = lr_spe + (size_t)b * 256 * 128;
    const h16* g32_b = guide32 + (size_t)b * 1024 * 128;
    const h16* g16_b = guide16 + (size_t)b * 256 * 128;
    const h16* spa_b = hr_spa + (size_t)b * 4096 * 128;
    h16* l1_b_feat = l1feat + (size_t)b * 1024 * 128;

    // ---- level 1: feat 16x16 -> query 32x32, residual FFN ----
    k_gather<<<1024 * 4, 128, 0, stream>>>(spe_b, 16, 16, g32_b, g16_b, 16, 16,
                                           32, 32, Abuf);
    gemm(8, 4096, Abuf, 416, 13, pk_l1, l1_b, 256, preds, 256, nullptr, 16, 1);
    gemm(4, 8192, preds, 128, 4, pk_wg32_1, wg32_b1, 64, wgh, 64, nullptr, 4, 2);
    gemm(8, 8192, wgh, 64, 2, pk_wg32_2, wg32_b2, 128, wgo, 128, nullptr, 8, 0);
    k_softmax_recon<<<1024, 128, 0, stream>>>(preds, wgo, recon);
    gemm(4, 1024, recon, 128, 4, pk_ffn32_1, ffn32_b1, 64, ffnh, 64, nullptr, 4, 2);
    gemm(8, 1024, ffnh, 64, 2, pk_ffn32_2, ffn32_b2, 128, l1_b_feat, 128, recon, 8, 0);

    // ---- level 2: feat 32x32 -> query 64x64, no residual ----
    k_gather<<<4096 * 4, 128, 0, stream>>>(l1_b_feat, 32, 32, spa_b, g32_b, 32, 32,
                                           64, 64, Abuf);
    gemm(8, 16384, Abuf, 416, 13, pk_l2, l2_b, 256, preds, 256, nullptr, 16, 1);
    gemm(4, 32768, preds, 128, 4, pk_wg64_1, wg64_b1, 64, wgh, 64, nullptr, 4, 2);
    gemm(8, 32768, wgh, 64, 2, pk_wg64_2, wg64_b2, 128, wgo, 128, nullptr, 8, 0);
    k_softmax_recon<<<4096, 128, 0, stream>>>(preds, wgo, recon);
    gemm(4, 4096, recon, 128, 4, pk_ffn64_1, ffn64_b1, 64, ffnh, 64, nullptr, 4, 2);
    gemm(2, 4096, ffnh, 64, 2, pk_ffn64_2, ffn64_b2, 31, l2o, 32, nullptr, 2, 0);
    k_final<<<(31 * 4096 + 255) / 256, 256, 0, stream>>>(lms, l2o, (float*)d_out, b);
  }
}
